// MultiHeadedAttention_33870112096588
// MI455X (gfx1250) — matrix operand order repaired and checked by reading, not compiled, not run on hardware
//
#include <hip/hip_runtime.h>
#include <hip/hip_bf16.h>
#include <math.h>

typedef _Float16 f16;
typedef __attribute__((ext_vector_type(16))) _Float16 v16h;
typedef __attribute__((ext_vector_type(8)))  float    v8f;
typedef __attribute__((ext_vector_type(4)))  unsigned int u32x4;

#define B_  2
#define T_  2048
#define D_  1024
#define H_  16
#define DK_ 64
#define M_  (B_ * T_)   // 4096 rows

union V16 {
  v16h  v;
  u32x4 q[2];
};

__device__ __forceinline__ v8f wmma32(v16h a, v16h b, v8f c) {
  // D = A(16x32 f16) x B(32x16 f16) + C(16x16 f32)
  return __builtin_amdgcn_wmma_f32_16x16x32_f16(false, a, false, b, (short)0, c,
                                                false, false);
}

// ---------------------------------------------------------------------------
// prep:
//   z=0..2 : convert q/k/v fp32 -> f16 (row-major [M,D])
//   z=3..6 : pack weights into B-operand order:
//            Wpk[((kblk*64 + nblk)*32 + lane)*16 + h] = W[n*D + k]
//            with k = kblk*32 + (lane>>4)*16 + h, n = nblk*16 + (lane&15)
// ---------------------------------------------------------------------------
__global__ void prep_kernel(const float* __restrict__ q, const float* __restrict__ k,
                            const float* __restrict__ v,
                            const float* __restrict__ Wq, const float* __restrict__ Wk,
                            const float* __restrict__ Wv, const float* __restrict__ Wo,
                            f16* q16, f16* k16, f16* v16p,
                            f16* WqP, f16* WkP, f16* WvP, f16* WoP) {
  int z = blockIdx.z;
  size_t idx = (size_t)blockIdx.x * blockDim.x + threadIdx.x;
  if (z < 3) {
    const float* src = (z == 0) ? q : (z == 1) ? k : v;
    f16* dst = (z == 0) ? q16 : (z == 1) ? k16 : v16p;
    if (idx < (size_t)M_ * D_) dst[idx] = (f16)src[idx];
  } else {
    const float* W = (z == 3) ? Wq : (z == 4) ? Wk : (z == 5) ? Wv : Wo;
    f16* WP = (z == 3) ? WqP : (z == 4) ? WkP : (z == 5) ? WvP : WoP;
    if (idx < (size_t)D_ * D_) {
      int h    = (int)(idx & 15);
      int lane = (int)((idx >> 4) & 31);
      int rest = (int)(idx >> 9);
      int nblk = rest & 63;    // D/16 = 64 column blocks
      int kblk = rest >> 6;    // D/32 = 32 k blocks
      int kk = kblk * 32 + (lane >> 4) * 16 + h;
      int nn = nblk * 16 + (lane & 15);
      WP[idx] = (f16)W[(size_t)nn * D_ + kk];
    }
  }
}

// ---------------------------------------------------------------------------
// proj: O = X @ W^T + b.  One wave computes a 16x64 tile (4 WMMAs / K-step,
// one A load shared by 4 B tiles).  z selects q/k/v.  z==2 (V) writes its
// result pre-swizzled into PV B-operand order:
//   Vpk[bh][(((key>>5)*4 + dk>>4)*32 + lane)*16 + h],
//   key = (key>>5)*32 + (lane>>4)*16 + h, dk = (dk>>4)*16 + (lane&15)
// ---------------------------------------------------------------------------
__global__ void __launch_bounds__(32) proj_kernel(
    const f16* __restrict__ q16, const f16* __restrict__ k16, const f16* __restrict__ v16p,
    const f16* __restrict__ WqP, const f16* __restrict__ WkP, const f16* __restrict__ WvP,
    const float* __restrict__ bq, const float* __restrict__ bk, const float* __restrict__ bv,
    f16* Qh, f16* Kh, f16* Vpk) {
  int z = blockIdx.z;
  const f16* X      = (z == 0) ? q16 : (z == 1) ? k16 : v16p;
  const f16* WP     = (z == 0) ? WqP : (z == 1) ? WkP : WvP;
  const float* bias = (z == 0) ? bq : (z == 1) ? bk : bv;
  f16* O = (z == 0) ? Qh : (z == 1) ? Kh : Vpk;

  int lane = threadIdx.x;
  int n = lane & 15, lh = lane >> 4;
  int n0 = blockIdx.x * 64;
  int m0 = blockIdx.y * 16;

  v8f acc[4] = {};
  const u32x4* xp = (const u32x4*)(X + (size_t)(m0 + n) * D_);
  for (int k0 = 0; k0 < D_; k0 += 32) {
    V16 a;
    int kq = k0 >> 3;  // u32x4 units
    a.q[0] = xp[kq + 2 * lh];
    a.q[1] = xp[kq + 2 * lh + 1];
#pragma unroll
    for (int nb = 0; nb < 4; ++nb) {
      const u32x4* wp = (const u32x4*)(
          WP + ((size_t)((k0 >> 5) * 64 + (n0 >> 4) + nb) * 32 + lane) * 16);
      V16 bm;
      bm.q[0] = wp[0];
      bm.q[1] = wp[1];
      acc[nb] = wmma32(a.v, bm.v, acc[nb]);
    }
  }

#pragma unroll
  for (int nb = 0; nb < 4; ++nb) {
    int d = n0 + nb * 16 + n;
    float bval = bias[d];
    int hh = d >> 6, dk = d & (DK_ - 1);
#pragma unroll
    for (int r = 0; r < 8; ++r) {
      int Mrow = m0 + r + lh * 8;
      int b = Mrow >> 11, t = Mrow & (T_ - 1);
      f16 val = (f16)(acc[nb][r] + bval);
      if (z < 2) {
        O[(((size_t)(b * H_ + hh)) * T_ + t) * DK_ + dk] = val;
      } else {
        // pre-swizzle into PV B-operand order
        int kblk = t >> 5, lh2 = (t >> 4) & 1, h2 = t & 15;
        int nb2 = dk >> 4, n2 = dk & 15;
        int lane2 = lh2 * 16 + n2;
        O[(size_t)(b * H_ + hh) * T_ * DK_ +
          (((size_t)(kblk * 4 + nb2) * 32 + lane2) * 16 + h2)] = val;
      }
    }
  }
}

// ---------------------------------------------------------------------------
// attn: flash attention, one wave per (b, h, 16-query tile).
// ---------------------------------------------------------------------------
__global__ void __launch_bounds__(32) attn_kernel(
    const f16* __restrict__ Qh, const f16* __restrict__ Kh, const f16* __restrict__ Vpk,
    const int* __restrict__ mask, f16* Xattn) {
  __shared__ __attribute__((aligned(16))) f16 pbuf[16 * 32];

  int lane = threadIdx.x;
  int n = lane & 15, lh = lane >> 4;
  int qt = blockIdx.x, h = blockIdx.y, b = blockIdx.z;
  const f16* Qb = Qh + (size_t)(b * H_ + h) * T_ * DK_;
  const f16* Kb = Kh + (size_t)(b * H_ + h) * T_ * DK_;
  const f16* Vb = Vpk + (size_t)(b * H_ + h) * T_ * DK_;
  const int* mb = mask + (size_t)b * T_;

  // Q tile as two A-chunks (K = 0..31 and 32..63), resident in VGPRs
  V16 qa0, qa1;
  {
    const u32x4* ap = (const u32x4*)(Qb + (size_t)(qt * 16 + n) * DK_);
    qa0.q[0] = ap[2 * lh];
    qa0.q[1] = ap[2 * lh + 1];
    qa1.q[0] = ap[4 + 2 * lh];
    qa1.q[1] = ap[5 + 2 * lh];
  }

  float rowm[8], rowl[8];
#pragma unroll
  for (int r = 0; r < 8; ++r) { rowm[r] = -1e30f; rowl[r] = 0.f; }
  v8f o[4] = {};
  const float scale = 0.125f;  // 1/sqrt(DK)

  for (int kb = 0; kb < T_; kb += 32) {
    // prefetch next key block's K rows and V block toward the WGP
    if (kb + 32 < T_) {
      __builtin_prefetch(Kb + (size_t)(kb + 32 + n) * DK_ + lh * 16, 0, 3);
      __builtin_prefetch(Kb + (size_t)(kb + 48 + n) * DK_ + lh * 16, 0, 3);
      __builtin_prefetch(
          Vb + ((((size_t)(kb >> 5) + 1) * 4) * 32 + lane) * 16, 0, 3);
    }

    float sv[2][8];
    float tmax[8];
#pragma unroll
    for (int r = 0; r < 8; ++r) tmax[r] = -INFINITY;

#pragma unroll
    for (int nb = 0; nb < 2; ++nb) {
      // B-operand: contiguous 32B per lane from K row (kb+nb*16+n)
      const u32x4* bp =
          (const u32x4*)(Kb + (size_t)(kb + nb * 16 + n) * DK_ + lh * 16);
      V16 bk0, bk1;
      bk0.q[0] = bp[0];
      bk0.q[1] = bp[1];
      bk1.q[0] = bp[4];   // +32 f16 -> kk = 32..63
      bk1.q[1] = bp[5];

      v8f s = {};
      s = wmma32(qa0.v, bk0.v, s);
      s = wmma32(qa1.v, bk1.v, s);

      int key = kb + nb * 16 + n;
      bool valid = (mb[key] != 0);
#pragma unroll
      for (int r = 0; r < 8; ++r) sv[nb][r] = valid ? s[r] * scale : -INFINITY;
#pragma unroll
      for (int r = 0; r < 8; ++r) {
        float x = sv[nb][r];
#pragma unroll
        for (int m2 = 1; m2 < 16; m2 <<= 1) x = fmaxf(x, __shfl_xor(x, m2, 32));
        tmax[r] = fmaxf(tmax[r], x);
      }
    }

    float fct[8];
#pragma unroll
    for (int r = 0; r < 8; ++r) {
      float nm = fmaxf(fmaxf(rowm[r], tmax[r]), -1e30f);
      fct[r] = __expf(rowm[r] - nm);
      rowm[r] = nm;
    }
#pragma unroll
    for (int r = 0; r < 8; ++r) {
      o[0][r] *= fct[r]; o[1][r] *= fct[r]; o[2][r] *= fct[r]; o[3][r] *= fct[r];
    }
    // probabilities -> LDS (row-major 16x32) + row-sum update
#pragma unroll
    for (int nb = 0; nb < 2; ++nb)
#pragma unroll
      for (int r = 0; r < 8; ++r) {
        float p = __expf(sv[nb][r] - rowm[r]);
        sv[nb][r] = p;
        pbuf[(r + lh * 8) * 32 + nb * 16 + n] = (f16)p;
      }
#pragma unroll
    for (int r = 0; r < 8; ++r) {
      float ps = sv[0][r] + sv[1][r];
#pragma unroll
      for (int m2 = 1; m2 < 16; m2 <<= 1) ps += __shfl_xor(ps, m2, 32);
      rowl[r] = rowl[r] * fct[r] + ps;
    }
    __syncthreads();  // single-wave WG: hardware NOP, compiler LDS fence

    // P back out of LDS in A-matrix layout (two ds b128 reads per lane)
    V16 pa;
    {
      const u32x4* pp = (const u32x4*)(pbuf + n * 32);
      pa.q[0] = pp[2 * lh];
      pa.q[1] = pp[2 * lh + 1];
    }
    // O += P @ V over 4 dk-blocks (V pre-swizzled: contiguous 32B per lane)
    const u32x4* vbase =
        (const u32x4*)(Vb + (((size_t)(kb >> 5) * 4) * 32 + lane) * 16);
#pragma unroll
    for (int nb2 = 0; nb2 < 4; ++nb2) {
      V16 bv_;
      bv_.q[0] = vbase[nb2 * 64 + 0];   // each dk-block: 32 lanes * 2 u32x4
      bv_.q[1] = vbase[nb2 * 64 + 1];
      o[nb2] = wmma32(pa.v, bv_.v, o[nb2]);
    }
    __syncthreads();
  }

  // normalize + store to [B,T,D] f16 (heads re-interleaved)
#pragma unroll
  for (int r = 0; r < 8; ++r) {
    float inv = (rowl[r] > 0.f) ? 1.f / rowl[r] : 0.f;
    int t = qt * 16 + r + lh * 8;
    f16* xp = Xattn + ((size_t)(b * T_ + t)) * D_ + h * DK_;
#pragma unroll
    for (int nb2 = 0; nb2 < 4; ++nb2) xp[nb2 * 16 + n] = (f16)(o[nb2][r] * inv);
  }
}

// ---------------------------------------------------------------------------
// outproj: out = Xattn @ Wo^T + bo, 16x64 tile per wave, fp32 result
// ---------------------------------------------------------------------------
__global__ void __launch_bounds__(32) outproj_kernel(
    const f16* __restrict__ X, const f16* __restrict__ WoP,
    const float* __restrict__ bo, float* __restrict__ out) {
  int lane = threadIdx.x;
  int n = lane & 15, lh = lane >> 4;
  int n0 = blockIdx.x * 64;
  int m0 = blockIdx.y * 16;

  v8f acc[4] = {};
  const u32x4* xp = (const u32x4*)(X + (size_t)(m0 + n) * D_);
  for (int k0 = 0; k0 < D_; k0 += 32) {
    V16 a;
    int kq = k0 >> 3;
    a.q[0] = xp[kq + 2 * lh];
    a.q[1] = xp[kq + 2 * lh + 1];
#pragma unroll
    for (int nb = 0; nb < 4; ++nb) {
      const u32x4* wp = (const u32x4*)(
          WoP + ((size_t)((k0 >> 5) * 64 + (n0 >> 4) + nb) * 32 + lane) * 16);
      V16 bm;
      bm.q[0] = wp[0];
      bm.q[1] = wp[1];
      acc[nb] = wmma32(a.v, bm.v, acc[nb]);
    }
  }
#pragma unroll
  for (int nb = 0; nb < 4; ++nb) {
    float bval = bo[n0 + nb * 16 + n];
#pragma unroll
    for (int r = 0; r < 8; ++r)
      out[(size_t)(m0 + r + lh * 8) * D_ + n0 + nb * 16 + n] = acc[nb][r] + bval;
  }
}

// ---------------------------------------------------------------------------
extern "C" void kernel_launch(void* const* d_in, const int* in_sizes, int n_in,
                              void* d_out, int out_size, void* d_ws, size_t ws_size,
                              hipStream_t stream) {
  const float* q    = (const float*)d_in[0];
  const float* k    = (const float*)d_in[1];
  const float* v    = (const float*)d_in[2];
  const int*   mask = (const int*)  d_in[3];
  const float* Wq   = (const float*)d_in[4];
  const float* bq   = (const float*)d_in[5];
  const float* Wk   = (const float*)d_in[6];
  const float* bk   = (const float*)d_in[7];
  const float* Wv   = (const float*)d_in[8];
  const float* bv   = (const float*)d_in[9];
  const float* Wo   = (const float*)d_in[10];
  const float* bo   = (const float*)d_in[11];
  float* out = (float*)d_out;

  size_t off = 0;
  auto take = [&](size_t bytes) {
    void* p = (char*)d_ws + off;
    off += (bytes + 255) & ~(size_t)255;
    return p;
  };
  f16* q16  = (f16*)take((size_t)M_ * D_ * sizeof(f16));
  f16* k16  = (f16*)take((size_t)M_ * D_ * sizeof(f16));
  f16* v16p = (f16*)take((size_t)M_ * D_ * sizeof(f16));
  f16* WqP  = (f16*)take((size_t)D_ * D_ * sizeof(f16));
  f16* WkP  = (f16*)take((size_t)D_ * D_ * sizeof(f16));
  f16* WvP  = (f16*)take((size_t)D_ * D_ * sizeof(f16));
  f16* WoP  = (f16*)take((size_t)D_ * D_ * sizeof(f16));
  f16* Qh   = (f16*)take((size_t)M_ * D_ * sizeof(f16));
  f16* Kh   = (f16*)take((size_t)M_ * D_ * sizeof(f16));
  f16* Vpk  = (f16*)take((size_t)M_ * D_ * sizeof(f16));
  f16* Xatt = (f16*)take((size_t)M_ * D_ * sizeof(f16));

  // 1) convert + weight pre-swizzle
  {
    dim3 grid(((size_t)M_ * D_ + 255) / 256, 1, 7);
    prep_kernel<<<grid, 256, 0, stream>>>(q, k, v, Wq, Wk, Wv, Wo,
                                          q16, k16, v16p, WqP, WkP, WvP, WoP);
  }
  // 2) q/k/v projections (V written pre-swizzled)
  {
    dim3 grid(D_ / 64, M_ / 16, 3);
    proj_kernel<<<grid, 32, 0, stream>>>(q16, k16, v16p, WqP, WkP, WvP,
                                         bq, bk, bv, Qh, Kh, Vpk);
  }
  // 3) flash attention
  {
    dim3 grid(T_ / 16, H_, B_);
    attn_kernel<<<grid, 32, 0, stream>>>(Qh, Kh, Vpk, mask, Xatt);
  }
  // 4) output projection
  {
    dim3 grid(D_ / 64, M_ / 16);
    outproj_kernel<<<grid, 32, 0, stream>>>(Xatt, WoP, bo, out);
  }
}
